// HierarchicalGenerator1_82987358093739
// MI455X (gfx1250) — matrix operand order repaired and checked by reading, not compiled, not run on hardware
//
#include <hip/hip_runtime.h>

// ---------------- problem constants ----------------
#define B_   256
#define T_   256
#define ZS_  128
#define ZT_  64
#define H_   512
#define G3_  1536          // 3*H
#define HPAD_ 520          // LDS row stride (bf16 elems), keeps 16B alignment
#define ZPAD_ 72
#define TAU_ 1.0f

typedef __attribute__((ext_vector_type(16))) __bf16 v16bf;
typedef __attribute__((ext_vector_type(8)))  float  v8f;
typedef unsigned short ushort_t;
typedef unsigned int   uint_t;

// ---------------- bf16 helpers ----------------
__device__ __forceinline__ ushort_t f2bf(float f) {
  union { float f; uint_t u; } x; x.f = f;
  uint_t u = x.u;
  uint_t r = (u + 0x7FFFu + ((u >> 16) & 1u)) >> 16;   // RNE
  return (ushort_t)r;
}
__device__ __forceinline__ float bf2f(ushort_t h) {
  union { uint_t u; float f; } x; x.u = ((uint_t)h) << 16; return x.f;
}
__device__ __forceinline__ float sigf(float x) { return 1.0f / (1.0f + __expf(-x)); }

// ---------------- WMMA fragment helpers ----------------
// A fragment (16x32 bf16, M x K): lane L holds row m=L&15,
// elems 0..7  -> k = kt*32 + (L>=16?8:0) + 0..7
// elems 8..15 -> k = kt*32 + 16 + (L>=16?8:0) + 0..7
// `base` is the lane-adjusted pointer (row m, +kA8); kt becomes an immediate.
__device__ __forceinline__ v16bf ldsA(const ushort_t* base, int kt) {
  union { uint4 q[2]; v16bf v; } u;
  u.q[0] = *(const uint4*)(base + kt * 32);
  u.q[1] = *(const uint4*)(base + kt * 32 + 16);
  return u.v;
}
// B fragment from packed weights: lane L holds col n=L&15,
// elem e -> k = kt*32 + (L>=16?8:0) + (e&7) + (e>=8?16:0).  Packed contiguously: 32B per lane.
// `bbase` hoists the (nt,lane) part so kt folds into the load immediate.
__device__ __forceinline__ const ushort_t* bbase(const ushort_t* __restrict__ wp,
                                                 int nt, int nkt, int lane) {
  return wp + ((size_t)nt * nkt * 32 + lane) * 16;
}
__device__ __forceinline__ v16bf gB(const ushort_t* base, int kt) {
  const uint4* p = (const uint4*)(base + kt * 512);
  union { uint4 q[2]; v16bf v; } u;
  u.q[0] = p[0]; u.q[1] = p[1];
  return u.v;
}
__device__ __forceinline__ v8f wmma_bf16(v16bf a, v16bf b, v8f c) {
  return __builtin_amdgcn_wmma_f32_16x16x32_bf16(false, a, false, b,
                                                 (short)0, c, false, false);
}

// ---------------- hash RNG / gumbel ----------------
__device__ __forceinline__ float urand(uint_t a, uint_t b, uint_t c) {
  uint_t x = a * 0x9E3779B1u + b * 0x85EBCA77u + c * 0xC2B2AE3Du + 0x27D4EB2Fu;
  x ^= x >> 16; x *= 0x7FEB352Du; x ^= x >> 15; x *= 0x846CA68Bu; x ^= x >> 16;
  return ((float)(x >> 8) + 0.5f) * (1.0f / 16777216.0f);
}
__device__ __forceinline__ float gumbelf(float u) { return -__logf(-__logf(u)); }

// ---------------- weight pack: W[N][ldw] f32 -> WMMA B tiles bf16 ----------------
__global__ void packW(const float* __restrict__ W, int ldw, int col0,
                      ushort_t* __restrict__ dst, int N, int K) {
  int tid = blockIdx.x * blockDim.x + threadIdx.x;
  int total = (N / 16) * (K / 32) * 512;
  if (tid >= total) return;
  int e = tid & 15;
  int lane = (tid >> 4) & 31;
  int tile = tid >> 9;
  int nkt = K / 32;
  int nt = tile / nkt, kt = tile % nkt;
  int n = nt * 16 + (lane & 15);
  int k = kt * 32 + ((lane >= 16) ? 8 : 0) + (e & 7) + ((e >= 8) ? 16 : 0);
  dst[tid] = f2bf(W[(size_t)n * ldw + col0 + k]);
}

// ---------------- gather 60 head rows into one [64][512] matrix + bias ----------
__global__ void gather_head(const float* tw, const float* tb,
                            const float* vw, const float* vb,
                            const float* tcw, const float* tcb,
                            const float* i0w, const float* i0b,
                            const float* i1w, const float* i1b,
                            const float* c0w, const float* c0b,
                            const float* c1w, const float* c1b,
                            float* __restrict__ HW, float* __restrict__ hb) {
  int tid = blockIdx.x * blockDim.x + threadIdx.x;
  if (tid >= 64 * H_) return;
  int r = tid >> 9, k = tid & (H_ - 1);
  float v = 0.f, bv = 0.f;
  if (r == 0)       { v = tw[k];                bv = tb[0]; }
  else if (r == 1)  { v = vw[k];                bv = vb[0]; }
  else if (r < 34)  { v = tcw[(r - 2) * H_ + k];  bv = tcb[r - 2]; }
  else if (r == 34) { v = i0w[k];               bv = i0b[0]; }
  else if (r == 35) { v = i1w[k];               bv = i1b[0]; }
  else if (r < 44)  { v = c0w[(r - 36) * H_ + k]; bv = c0b[r - 36]; }
  else if (r < 60)  { v = c1w[(r - 44) * H_ + k]; bv = c1b[r - 44]; }
  HW[tid] = v;
  if (k == 0) hb[r] = bv;
}

// ---------------- generic WMMA GEMM: C = act(A @ Wpackᵀ + bias) ----------------
template <int K>
__global__ __launch_bounds__(256, 2)
void gemm16(const float* __restrict__ A,
            const ushort_t* __restrict__ Bp, int N,
            const float* __restrict__ bias,
            float* __restrict__ C, int ldc, int relu) {
  __shared__ ushort_t As[16 * HPAD_];
  int tid = threadIdx.x, lane = tid & 31, w = tid >> 5;
  int m0 = blockIdx.x * 16;
  for (int i = tid; i < 16 * K; i += 256) {
    int m = i / K, k = i - m * K;
    As[m * HPAD_ + k] = f2bf(A[(size_t)(m0 + m) * K + k]);
  }
  __syncthreads();
  int nt = blockIdx.y * 8 + w;
  if (nt * 16 >= N) return;
  int nD = lane & 15, hi8 = (lane >= 16) ? 8 : 0;
  constexpr int nkt = K / 32;
  const ushort_t* ab = As + nD * HPAD_ + hi8;
  const ushort_t* bb_ = bbase(Bp, nt, nkt, lane);
  v8f acc;
  float bv = bias[nt * 16 + nD];
  for (int r = 0; r < 8; r++) acc[r] = bv;
#pragma unroll
  for (int kt = 0; kt < nkt; kt++)
    acc = wmma_bf16(ldsA(ab, kt), gB(bb_, kt), acc);
  for (int r = 0; r < 8; r++) {
    float v = acc[r];
    if (relu) v = fmaxf(v, 0.f);
    C[(size_t)(m0 + r + hi8) * ldc + nt * 16 + nD] = v;
  }
}

// ---------------- small static heads: slin[B][32] = [scont(16) | scat(10)] ----
__global__ __launch_bounds__(32)
void static_heads(const float* __restrict__ sh,
                  const float* scw, const float* scb,
                  const float* saw, const float* sab,
                  float* __restrict__ slin) {
  int b = blockIdx.x, j = threadIdx.x;
  if (j >= 26) return;
  const float* wr; float s;
  if (j < 16) { wr = scw + j * H_;        s = scb[j]; }
  else        { wr = saw + (j - 16) * H_; s = sab[j - 16]; }
  const float* hb = sh + (size_t)b * H_;
  for (int k = 0; k < H_; k++) s += hb[k] * wr[k];
  slin[b * 32 + j] = s;
}

// ---------------- fused 2-layer GRU scan + head projection --------------------
// 16 workgroups x 256 threads (8 waves). WG owns 16 batch rows for all T.
// e0[B][1536] = b_ih0 + static_h @ w_ih0[:,64:]^T (precomputed, time-invariant).
__global__ __launch_bounds__(256, 1)
void gru_scan(const float* __restrict__ zt,       // [B,T,64]
              const float* __restrict__ e0,       // [B,1536]
              const float* __restrict__ bhh0,     // [1536]
              const float* __restrict__ bih1,     // [1536]
              const float* __restrict__ bhh1,     // [1536]
              const ushort_t* __restrict__ wih0zp,// packed N=1536 K=64
              const ushort_t* __restrict__ whh0p, // packed N=1536 K=512
              const ushort_t* __restrict__ wih1p,
              const ushort_t* __restrict__ whh1p,
              const ushort_t* __restrict__ headp, // packed N=64  K=512
              const float* __restrict__ headb,    // [64]
              float* __restrict__ hlin) {         // [B,T,64]
  __shared__ ushort_t h0s[16 * HPAD_];
  __shared__ ushort_t h1s[16 * HPAD_];
  __shared__ ushort_t zs[16 * ZPAD_];
  int tid = threadIdx.x, lane = tid & 31, w = tid >> 5;
  int b0 = blockIdx.x * 16;
  int nD = lane & 15;
  int hi8 = (lane >= 16) ? 8 : 0;
  const ushort_t* h0a = h0s + nD * HPAD_ + hi8;   // A-frag lane base (layer-0 h)
  const ushort_t* h1a = h1s + nD * HPAD_ + hi8;   // A-frag lane base (layer-1 h)
  const ushort_t* za  = zs + nD * ZPAD_ + hi8;    // A-frag lane base (z_t)
  const ushort_t* hwb = bbase(headp, w, 16, lane);
  for (int i = tid; i < 16 * HPAD_; i += 256) { h0s[i] = 0; h1s[i] = 0; }
  __syncthreads();

  for (int t = 0; t < T_; t++) {
    // stage z_t (f32 -> bf16) into LDS
    for (int i = tid; i < 16 * ZT_; i += 256) {
      int m = i >> 6, k = i & 63;
      zs[m * ZPAD_ + k] = f2bf(zt[((size_t)(b0 + m) * T_ + t) * ZT_ + k]);
    }
    __syncthreads();

    // ---- layer 0 ----
    v8f h0new[4];
    for (int c4 = 0; c4 < 4; c4++) {
      int c = w * 4 + c4;
      int colb = c * 16 + nD;
      const ushort_t* wzr = bbase(wih0zp,      c, 2, lane);
      const ushort_t* wzz = bbase(wih0zp, 32 + c, 2, lane);
      const ushort_t* wzn = bbase(wih0zp, 64 + c, 2, lane);
      const ushort_t* whr = bbase(whh0p,       c, 16, lane);
      const ushort_t* whz = bbase(whh0p,  32 + c, 16, lane);
      const ushort_t* whn = bbase(whh0p,  64 + c, 16, lane);
      v8f racc, zacc, xnacc, gnacc;
      float bhr = bhh0[colb], bhz = bhh0[H_ + colb], bhn = bhh0[2 * H_ + colb];
      for (int r = 0; r < 8; r++) {
        size_t row = (size_t)(b0 + hi8 + r) * G3_;
        racc[r]  = e0[row + colb] + bhr;
        zacc[r]  = e0[row + H_ + colb] + bhz;
        xnacc[r] = e0[row + 2 * H_ + colb];
        gnacc[r] = bhn;
      }
#pragma unroll
      for (int kt = 0; kt < 2; kt++) {        // z_t input part (K=64)
        v16bf a = ldsA(za, kt);
        racc  = wmma_bf16(a, gB(wzr, kt), racc);
        zacc  = wmma_bf16(a, gB(wzz, kt), zacc);
        xnacc = wmma_bf16(a, gB(wzn, kt), xnacc);
      }
#pragma unroll 4
      for (int kt = 0; kt < 16; kt++) {       // recurrent part (K=512)
        v16bf a = ldsA(h0a, kt);
        racc  = wmma_bf16(a, gB(whr, kt), racc);
        zacc  = wmma_bf16(a, gB(whz, kt), zacc);
        gnacc = wmma_bf16(a, gB(whn, kt), gnacc);
      }
      for (int r = 0; r < 8; r++) {
        float rr = sigf(racc[r]);
        float zz = sigf(zacc[r]);
        float nn = tanhf(xnacc[r] + rr * gnacc[r]);
        float hold = bf2f(h0s[(r + hi8) * HPAD_ + colb]);
        h0new[c4][r] = (1.f - zz) * nn + zz * hold;
      }
    }
    __syncthreads();
    for (int c4 = 0; c4 < 4; c4++) {
      int colb = (w * 4 + c4) * 16 + nD;
      for (int r = 0; r < 8; r++) h0s[(r + hi8) * HPAD_ + colb] = f2bf(h0new[c4][r]);
    }
    __syncthreads();

    // ---- layer 1 (input = h0s at time t) ----
    v8f h1new[4];
    for (int c4 = 0; c4 < 4; c4++) {
      int c = w * 4 + c4;
      int colb = c * 16 + nD;
      const ushort_t* wxr = bbase(wih1p,      c, 16, lane);
      const ushort_t* wxz = bbase(wih1p, 32 + c, 16, lane);
      const ushort_t* wxn = bbase(wih1p, 64 + c, 16, lane);
      const ushort_t* whr = bbase(whh1p,      c, 16, lane);
      const ushort_t* whz = bbase(whh1p, 32 + c, 16, lane);
      const ushort_t* whn = bbase(whh1p, 64 + c, 16, lane);
      v8f racc, zacc, xnacc, gnacc;
      float br  = bih1[colb] + bhh1[colb];
      float bz  = bih1[H_ + colb] + bhh1[H_ + colb];
      float bxn = bih1[2 * H_ + colb];
      float bgn = bhh1[2 * H_ + colb];
      for (int r = 0; r < 8; r++) { racc[r] = br; zacc[r] = bz; xnacc[r] = bxn; gnacc[r] = bgn; }
#pragma unroll 4
      for (int kt = 0; kt < 16; kt++) {
        v16bf ax = ldsA(h0a, kt);
        racc  = wmma_bf16(ax, gB(wxr, kt), racc);
        zacc  = wmma_bf16(ax, gB(wxz, kt), zacc);
        xnacc = wmma_bf16(ax, gB(wxn, kt), xnacc);
        v16bf ah = ldsA(h1a, kt);
        racc  = wmma_bf16(ah, gB(whr, kt), racc);
        zacc  = wmma_bf16(ah, gB(whz, kt), zacc);
        gnacc = wmma_bf16(ah, gB(whn, kt), gnacc);
      }
      for (int r = 0; r < 8; r++) {
        float rr = sigf(racc[r]);
        float zz = sigf(zacc[r]);
        float nn = tanhf(xnacc[r] + rr * gnacc[r]);
        float hold = bf2f(h1s[(r + hi8) * HPAD_ + colb]);
        h1new[c4][r] = (1.f - zz) * nn + zz * hold;
      }
    }
    __syncthreads();
    for (int c4 = 0; c4 < 4; c4++) {
      int colb = (w * 4 + c4) * 16 + nD;
      for (int r = 0; r < 8; r++) h1s[(r + hi8) * HPAD_ + colb] = f2bf(h1new[c4][r]);
    }
    __syncthreads();

    // ---- fused head projection: hlin[b][t][0..63] = h1_t @ Wheadᵀ + bhead ----
    if (w < 4) {
      v8f acc;
      float bb = headb[w * 16 + nD];
      for (int r = 0; r < 8; r++) acc[r] = bb;
#pragma unroll 4
      for (int kt = 0; kt < 16; kt++)
        acc = wmma_bf16(ldsA(h1a, kt), gB(hwb, kt), acc);
      for (int r = 0; r < 8; r++)
        hlin[((size_t)(b0 + hi8 + r) * T_ + t) * 64 + w * 16 + nD] = acc[r];
    }
    if (t + 1 < T_ && tid < 16)
      __builtin_prefetch(&zt[((size_t)(b0 + tid) * T_ + (t + 1)) * ZT_], 0, 0);
  }
}

// ---------------- argmax over wave (inactive lanes pass -inf) -----------------
__device__ __forceinline__ int wave_argmax(float v, int idx) {
  for (int off = 16; off > 0; off >>= 1) {
    float ov = __shfl_xor(v, off, 32);
    int   oi = __shfl_xor(idx, off, 32);
    if (ov > v || (ov == v && oi < idx)) { v = ov; idx = oi; }
  }
  return idx;
}

// ---------------- sequential elementwise heads: one wave per batch row --------
__global__ __launch_bounds__(32)
void heads_kernel(const float* __restrict__ slin,   // [B,32]
                  const float* __restrict__ hlin,   // [B,T,64]
                  float* __restrict__ out) {
  int b = blockIdx.x, lane = threadIdx.x;
  float* O0 = out;                 // static_cont   [B,16]
  float* O1 = out + 4096;          // static_cat    [B,10]
  float* O2 = out + 6656;          // temporal_cont [B,T,32]
  float* O3 = out + 2103808;       // irr0          [B,T,2]
  float* O4 = out + 2234880;       // irr1          [B,T,2]
  float* O5 = out + 2365952;       // cat0          [B,T,8]
  float* O6 = out + 2890240;       // cat1          [B,T,16]
  float* O7 = out + 3938816;       // visit_mask    [B,T,1]
  float* O8 = out + 4004352;       // visit_times   [B,T]

  if (lane < 16) O0[b * 16 + lane] = slin[b * 32 + lane];
  {
    float v = -1e30f;
    if (lane < 10)
      v = (slin[b * 32 + 16 + lane] + gumbelf(urand(b, 9999u, 100u + lane))) / TAU_;
    int am = wave_argmax(v, lane);
    if (lane < 10) O1[b * 10 + lane] = (lane == am) ? 1.0f : 0.0f;
  }

  float mask = 1.0f, cum = 0.0f, prev0 = 0.0f, prev1 = 0.0f;
  for (int t = 0; t < T_; t++) {
    const float* hl = hlin + ((size_t)(b * T_ + t)) * 64;
    float d = hl[0];
    float delta = (d > 20.f) ? d : log1pf(__expf(d));   // softplus
    cum += delta;
    float vp = sigf(hl[1]);
    float vm = (t == 0) ? 1.f : ((urand(b, t, 1u) < vp) ? 1.f : 0.f);
    mask *= vm;

    O2[((size_t)(b * T_ + t)) * 32 + lane] = hl[2 + lane] * mask;

    float hz0 = sigf(hl[34]);
    float fl0 = (urand(b, t, 2u) < hz0 * (1.f - prev0)) ? 1.f : 0.f; prev0 += fl0;
    float hz1 = sigf(hl[35]);
    float fl1 = (urand(b, t, 3u) < hz1 * (1.f - prev1)) ? 1.f : 0.f; prev1 += fl1;
    if (lane == 0) {
      size_t p = (size_t)(b * T_ + t) * 2;
      O3[p] = (1.f - prev0) * mask; O3[p + 1] = prev0 * mask;
      O4[p] = (1.f - prev1) * mask; O4[p + 1] = prev1 * mask;
      O7[b * T_ + t] = mask;
      O8[b * T_ + t] = cum;
    }
    {
      float v = -1e30f;
      if (lane < 8) v = (hl[36 + lane] + gumbelf(urand(b, t, 10u + lane))) / TAU_;
      int am = wave_argmax(v, lane);
      if (lane < 8) O5[(size_t)(b * T_ + t) * 8 + lane] = ((lane == am) ? 1.f : 0.f) * mask;
    }
    {
      float v = -1e30f;
      if (lane < 16) v = (hl[44 + lane] + gumbelf(urand(b, t, 40u + lane))) / TAU_;
      int am = wave_argmax(v, lane);
      if (lane < 16) O6[(size_t)(b * T_ + t) * 16 + lane] = ((lane == am) ? 1.f : 0.f) * mask;
    }
  }
  float scale = 1.f / (cum + 1e-8f);
  for (int t = lane; t < T_; t += 32) O8[b * T_ + t] *= scale;
}

// ---------------- launcher ----------------
extern "C" void kernel_launch(void* const* d_in, const int* in_sizes, int n_in,
                              void* d_out, int out_size, void* d_ws, size_t ws_size,
                              hipStream_t stream) {
  (void)in_sizes; (void)n_in; (void)out_size; (void)ws_size;
  const float* z_static   = (const float*)d_in[0];
  const float* z_temporal = (const float*)d_in[1];
  const float* fc1_w = (const float*)d_in[2];  const float* fc1_b = (const float*)d_in[3];
  const float* fc2_w = (const float*)d_in[4];  const float* fc2_b = (const float*)d_in[5];
  const float* w_ih0 = (const float*)d_in[6];  const float* w_hh0 = (const float*)d_in[7];
  const float* b_ih0 = (const float*)d_in[8];  const float* b_hh0 = (const float*)d_in[9];
  const float* w_ih1 = (const float*)d_in[10]; const float* w_hh1 = (const float*)d_in[11];
  const float* b_ih1 = (const float*)d_in[12]; const float* b_hh1 = (const float*)d_in[13];
  const float* time_w = (const float*)d_in[14];  const float* time_b = (const float*)d_in[15];
  const float* visit_w = (const float*)d_in[16]; const float* visit_b = (const float*)d_in[17];
  const float* tcont_w = (const float*)d_in[18]; const float* tcont_b = (const float*)d_in[19];
  const float* irr0_w = (const float*)d_in[20];  const float* irr0_b = (const float*)d_in[21];
  const float* irr1_w = (const float*)d_in[22];  const float* irr1_b = (const float*)d_in[23];
  const float* cat0_w = (const float*)d_in[24];  const float* cat0_b = (const float*)d_in[25];
  const float* cat1_w = (const float*)d_in[26];  const float* cat1_b = (const float*)d_in[27];
  const float* scont_w = (const float*)d_in[28]; const float* scont_b = (const float*)d_in[29];
  const float* scat_w = (const float*)d_in[30];  const float* scat_b = (const float*)d_in[31];

  char* ws = (char*)d_ws;
  size_t off = 0;
  auto alloc = [&](size_t bytes) { size_t o = off; off += (bytes + 255) & ~(size_t)255; return o; };
  ushort_t* fc1p   = (ushort_t*)(ws + alloc((size_t)512 * 128 * 2));
  ushort_t* fc2p   = (ushort_t*)(ws + alloc((size_t)512 * 512 * 2));
  ushort_t* wih0zp = (ushort_t*)(ws + alloc((size_t)1536 * 64 * 2));
  ushort_t* wih0sp = (ushort_t*)(ws + alloc((size_t)1536 * 512 * 2));
  ushort_t* whh0p  = (ushort_t*)(ws + alloc((size_t)1536 * 512 * 2));
  ushort_t* wih1p  = (ushort_t*)(ws + alloc((size_t)1536 * 512 * 2));
  ushort_t* whh1p  = (ushort_t*)(ws + alloc((size_t)1536 * 512 * 2));
  ushort_t* headp  = (ushort_t*)(ws + alloc((size_t)64 * 512 * 2));
  float* headHW = (float*)(ws + alloc((size_t)64 * 512 * 4));
  float* headhb = (float*)(ws + alloc((size_t)64 * 4));
  float* t1      = (float*)(ws + alloc((size_t)256 * 512 * 4));
  float* statich = (float*)(ws + alloc((size_t)256 * 512 * 4));
  float* e0      = (float*)(ws + alloc((size_t)256 * 1536 * 4));
  float* slin    = (float*)(ws + alloc((size_t)256 * 32 * 4));
  float* hlin    = (float*)(ws + alloc((size_t)256 * 256 * 64 * 4));

  // 1) pack weights to WMMA bf16 fragment layout
  packW<<<256, 256, 0, stream>>>(fc1_w, 128, 0, fc1p, 512, 128);
  packW<<<1024, 256, 0, stream>>>(fc2_w, 512, 0, fc2p, 512, 512);
  packW<<<384, 256, 0, stream>>>(w_ih0, 576, 0, wih0zp, 1536, 64);    // z_t slice
  packW<<<3072, 256, 0, stream>>>(w_ih0, 576, 64, wih0sp, 1536, 512); // static slice
  packW<<<3072, 256, 0, stream>>>(w_hh0, 512, 0, whh0p, 1536, 512);
  packW<<<3072, 256, 0, stream>>>(w_ih1, 512, 0, wih1p, 1536, 512);
  packW<<<3072, 256, 0, stream>>>(w_hh1, 512, 0, whh1p, 1536, 512);
  gather_head<<<128, 256, 0, stream>>>(time_w, time_b, visit_w, visit_b,
                                       tcont_w, tcont_b, irr0_w, irr0_b,
                                       irr1_w, irr1_b, cat0_w, cat0_b,
                                       cat1_w, cat1_b, headHW, headhb);
  packW<<<128, 256, 0, stream>>>(headHW, 512, 0, headp, 64, 512);

  // 2) static MLP + time-invariant layer-0 input projection
  gemm16<128><<<dim3(16, 4), 256, 0, stream>>>(z_static, fc1p, 512, fc1_b, t1, 512, 1);
  gemm16<512><<<dim3(16, 4), 256, 0, stream>>>(t1, fc2p, 512, fc2_b, statich, 512, 0);
  gemm16<512><<<dim3(16, 12), 256, 0, stream>>>(statich, wih0sp, 1536, b_ih0, e0, 1536, 0);
  static_heads<<<256, 32, 0, stream>>>(statich, scont_w, scont_b, scat_w, scat_b, slin);

  // 3) fused two-layer GRU scan + head projection (all WMMA, states in LDS)
  gru_scan<<<16, 256, 0, stream>>>(z_temporal, e0, b_hh0, b_ih1, b_hh1,
                                   wih0zp, whh0p, wih1p, whh1p, headp, headhb, hlin);

  // 4) sequential elementwise heads
  heads_kernel<<<256, 32, 0, stream>>>(slin, hlin, (float*)d_out);
}
